// QuantLinear_29832842838126
// MI455X (gfx1250) — matrix operand order repaired and checked by reading, not compiled, not run on hardware
//
#include <hip/hip_runtime.h>

// ---------------------------------------------------------------------------
// W4A16 quantized linear for MI455X (gfx1250, wave32, WMMA, async-to-LDS).
//
//   Y[2048,4096] = X[2048,4096] @ dequant(qweight)[4096,4096] + bias
//
// 3-pass design (all intermediates fit in the 192MB L2):
//   pass 1: X fp32 -> f16 into ws (once, instead of per-N-block)
//   pass 2: qweight -> f16 W, pre-transposed [N][K] into ws (once, fp32
//           dequant math, single rounding; instead of per-M-block)
//   pass 3: f16 x f16 GEMM, fp32 accum, v_wmma_f32_16x16x32_f16.
//           Tiles staged with global_load_async_to_lds_b128 (ASYNCcnt),
//           double-buffered LDS, software pipelined (s_wait_asynccnt),
//           last iteration peeled so the hot loop is branchless.
//
// Pass-3 tiling: 128(M) x 256(N) block, 8 waves as 2(M) x 4(N), each wave a
// 64x64 tile = 4x4 fragments -> 16 WMMAs per 16 ds_load_b128 per iteration.
// All 8 fragments (4 A + 4 B) kept resident so LDS loads pipeline without
// full s_wait_dscnt drains between WMMA groups.
//
// Workspace requirement: (tokens*INF + INF*OUTF)*2 bytes = 48 MB.
// ---------------------------------------------------------------------------

typedef __attribute__((ext_vector_type(16))) _Float16 v16h;
typedef __attribute__((ext_vector_type(8)))  _Float16 v8h;
typedef __attribute__((ext_vector_type(8)))  float    v8f;
typedef __attribute__((ext_vector_type(4)))  float    f32x4;

#define INF    4096      // infeatures (K)
#define OUTF   4096      // outfeatures (N), must stay 4096 (we use >>12)
#define QGROUP 128       // quant group size along K
#define BM 128           // block tile M
#define BN 256           // block tile N
#define BK 32            // k step = WMMA K depth (f16)
#define KTILES (INF / BK)
#define LDX 40           // padded f16 row stride, 80B rows -> 16B aligned chunks
#define LDW 40

// ---------------------------------------------------------------------------
// Pass 1: X fp32 -> f16 (8 elements / thread)
// ---------------------------------------------------------------------------
__global__ __launch_bounds__(256)
void convert_x_kernel(const float* __restrict__ x, _Float16* __restrict__ xh)
{
    const size_t i = ((size_t)blockIdx.x * 256 + threadIdx.x) * 8;
    f32x4 a = *(const f32x4*)(x + i);
    f32x4 b = *(const f32x4*)(x + i + 4);
    v8h h;
    h[0] = (_Float16)a[0]; h[1] = (_Float16)a[1];
    h[2] = (_Float16)a[2]; h[3] = (_Float16)a[3];
    h[4] = (_Float16)b[0]; h[5] = (_Float16)b[1];
    h[6] = (_Float16)b[2]; h[7] = (_Float16)b[3];
    *(v8h*)(xh + i) = h;
}

// ---------------------------------------------------------------------------
// Pass 2: dequant qweight -> f16 W stored transposed [N][K].
// One thread per packed int32 (8 K-values of one column). Reads coalesced;
// each thread writes one contiguous 16B chunk wt[n][kq*8 .. +7].
// ---------------------------------------------------------------------------
__global__ __launch_bounds__(256)
void dequant_w_kernel(const int*   __restrict__ qw,
                      const float* __restrict__ scales,
                      const float* __restrict__ zeros,
                      _Float16*    __restrict__ wt)
{
    const int idx = blockIdx.x * 256 + threadIdx.x;   // 0 .. INF/8*OUTF-1
    const int n   = idx & (OUTF - 1);                 // column
    const int kq  = idx >> 12;                        // packed row (8 K each)
    const int g   = kq >> 4;                          // (kq*8)/QGROUP

    float s = scales[(size_t)g * OUTF + n];
    s = fminf(fmaxf(s, 1e-5f), 1e4f);                 // CLIPMIN/CLIPMAX
    float z = zeros[(size_t)g * OUTF + n];
    z = fminf(fmaxf(rintf(z), 0.0f), 15.0f);          // round-half-even + clamp

    const int q = qw[idx];
    v8h o;
#pragma unroll
    for (int j = 0; j < 8; ++j) {
        const float wq = (float)((q >> (4 * j)) & 15);
        o[j] = (_Float16)((wq - z) * s);              // fp32 math, one rounding
    }
    *(v8h*)(wt + (size_t)n * INF + kq * 8) = o;
}

// ---------------------------------------------------------------------------
// Pass 3: f16 GEMM with WMMA + async-to-LDS double-buffered pipeline.
// ---------------------------------------------------------------------------
__global__ __launch_bounds__(256)
void w4a16_wmma_main(const _Float16* __restrict__ xh,   // [tokens][INF] f16
                     const _Float16* __restrict__ wt,   // [OUTF][INF]  f16 (transposed)
                     const float*    __restrict__ bias,
                     float*          __restrict__ out)
{
    __shared__ __align__(16) _Float16 ldsX[2][BM * LDX];   // 2 x 10KB
    __shared__ __align__(16) _Float16 ldsW[2][BN * LDW];   // 2 x 20KB ([N][K])

    const int tid   = threadIdx.x;
    const int wave  = tid >> 5;
    const int lane  = tid & 31;
    const int lmod  = lane & 15;
    const int lhalf = lane >> 4;

    const int gm = blockIdx.y * BM;
    const int gn = blockIdx.x * BN;
    const int wm = (wave & 1) * 64;       // 2 waves along M
    const int wn = (wave >> 1) * 64;      // 4 waves along N

    v8f acc[4][4];
#pragma unroll
    for (int fm = 0; fm < 4; ++fm)
#pragma unroll
        for (int fn = 0; fn < 4; ++fn)
            acc[fm][fn] = (v8f)0.0f;

    // Stage k-tile t into LDS buffer b with async copies (ASYNCcnt).
    // X tile: 128x32 f16 = 512 16B chunks (i=0,1); W tile: 256x32 f16 =
    // 1024 chunks (i=0..3). 6 async b128s per thread per tile.
    auto stage = [&](int t, int b) {
        const int k0 = t * BK;
#pragma unroll
        for (int i = 0; i < 2; ++i) {
            const int c   = tid + i * 256;          // 0..511
            const int row = c >> 2;
            const int cq  = c & 3;
            unsigned lx = (unsigned)(unsigned long long)&ldsX[b][row * LDX + cq * 8];
            const _Float16* gx = xh + (size_t)(gm + row) * INF + k0 + cq * 8;
            asm volatile("global_load_async_to_lds_b128 %0, %1, off"
                         :: "v"(lx), "v"(gx) : "memory");
        }
#pragma unroll
        for (int i = 0; i < 4; ++i) {
            const int c   = tid + i * 256;          // 0..1023
            const int n   = c >> 2;
            const int cq  = c & 3;
            unsigned lw = (unsigned)(unsigned long long)&ldsW[b][n * LDW + cq * 8];
            const _Float16* gw = wt + (size_t)(gn + n) * INF + k0 + cq * 8;
            asm volatile("global_load_async_to_lds_b128 %0, %1, off"
                         :: "v"(lw), "v"(gw) : "memory");
        }
    };

    // One k-tile of compute out of LDS buffer `cur`. All 4 A and 4 B
    // fragments loaded into distinct registers first so the 16 ds_load_b128
    // can overlap and the scheduler can use partial dscnt waits.
    auto compute = [&](int cur) {
        // B (32x16 f16): lane = column lmod; contiguous K [lhalf*16, +16).
        v16h bfr[4];
#pragma unroll
        for (int fn = 0; fn < 4; ++fn) {
            const _Float16* p = &ldsW[cur][(wn + fn * 16 + lmod) * LDW + lhalf * 16];
            v8h lo = *(const v8h*)p;
            v8h hi = *(const v8h*)(p + 8);
            bfr[fn] = __builtin_shufflevector(lo, hi, 0, 1, 2, 3, 4, 5, 6, 7,
                                                      8, 9, 10, 11, 12, 13, 14, 15);
        }
        // A (16x32 f16): lane = row lmod; contiguous K [lhalf*16, +16).
        v16h afr[4];
#pragma unroll
        for (int fm = 0; fm < 4; ++fm) {
            const _Float16* p = &ldsX[cur][(wm + fm * 16 + lmod) * LDX + lhalf * 16];
            v8h lo = *(const v8h*)p;
            v8h hi = *(const v8h*)(p + 8);
            afr[fm] = __builtin_shufflevector(lo, hi, 0, 1, 2, 3, 4, 5, 6, 7,
                                                      8, 9, 10, 11, 12, 13, 14, 15);
        }
#pragma unroll
        for (int fm = 0; fm < 4; ++fm)
#pragma unroll
            for (int fn = 0; fn < 4; ++fn)
                acc[fm][fn] = __builtin_amdgcn_wmma_f32_16x16x32_f16(
                    false, afr[fm], false, bfr[fn],
                    (short)0, acc[fm][fn], false, false);
    };

    stage(0, 0);    // prologue: tile 0 in flight

    // Steady state: branchless body, last iteration peeled.
#pragma unroll 1
    for (int t = 0; t < KTILES - 1; ++t) {
        const int cur = t & 1;
        stage(t + 1, cur ^ 1);                       // next tile in flight
        // Async loads retire in order: <=6 outstanding => tile t landed,
        // tile t+1's 6 still overlap with this iteration's compute.
        asm volatile("s_wait_asynccnt 0x6" ::: "memory");
        __syncthreads();   // all waves' tile-t async writes visible
        compute(cur);
        __syncthreads();   // reads of buf cur done before tile t+2 lands in it
    }
    // Final tile (no further staging).
    asm volatile("s_wait_asynccnt 0x0" ::: "memory");
    __syncthreads();
    compute((KTILES - 1) & 1);

    // ---- epilogue: C/D layout (VGPR r -> row r / r+8 by lane half; col = lane%16)
#pragma unroll
    for (int fn = 0; fn < 4; ++fn) {
        const int col = gn + wn + fn * 16 + lmod;
        const float b = bias[col];
#pragma unroll
        for (int fm = 0; fm < 4; ++fm) {
            const int rbase = gm + wm + fm * 16 + lhalf * 8;
#pragma unroll
            for (int r = 0; r < 8; ++r)
                out[(size_t)(rbase + r) * OUTF + col] = acc[fm][fn][r] + b;
        }
    }
}

// ---------------------------------------------------------------------------
// Host launch. Inputs per setup_inputs(): x, scales, zero_points, bias, qweight.
// Deterministic: ws is fully rewritten by passes 1-2 before pass 3 reads it.
// ---------------------------------------------------------------------------
extern "C" void kernel_launch(void* const* d_in, const int* in_sizes, int n_in,
                              void* d_out, int out_size, void* d_ws, size_t ws_size,
                              hipStream_t stream)
{
    const float* x       = (const float*)d_in[0];
    const float* scales  = (const float*)d_in[1];
    const float* zeros   = (const float*)d_in[2];
    const float* bias    = (const float*)d_in[3];
    const int*   qweight = (const int*)d_in[4];
    float*       out     = (float*)d_out;

    const int tokens = in_sizes[0] / INF;          // 2048

    _Float16* xh = (_Float16*)d_ws;                       // [tokens][INF]
    _Float16* wt = xh + (size_t)tokens * INF;             // [OUTF][INF]

    // pass 1: X -> f16
    convert_x_kernel<<<(size_t)tokens * INF / (256 * 8), 256, 0, stream>>>(x, xh);
    // pass 2: dequant W -> f16, transposed
    dequant_w_kernel<<<(INF / 8) * OUTF / 256, 256, 0, stream>>>(qweight, scales, zeros, wt);
    // pass 3: WMMA GEMM
    dim3 grid(OUTF / BN, tokens / BM);             // 16 x 16
    w4a16_wmma_main<<<grid, dim3(256, 1, 1), 0, stream>>>(xh, wt, bias, out);
}
